// decoder_65558380806687
// MI455X (gfx1250) — matrix operand order repaired and checked by reading, not compiled, not run on hardware
//
#include <hip/hip_runtime.h>
#include <hip/hip_bf16.h>

typedef __attribute__((ext_vector_type(16))) _Float16 v16h;
typedef __attribute__((ext_vector_type(8)))  _Float16 v8h;
typedef __attribute__((ext_vector_type(8)))  float    v8f;
typedef _Float16 half_t;

#define NBLK 24          // 24 blocks * 8 waves = 192 waves = 4 x 48 h-tiles
#define BH   (64*768)    // one h buffer (elements)

// ---------------- math helpers ----------------
__device__ __forceinline__ float sigm(float x) { return 1.0f / (1.0f + __expf(-x)); }
__device__ __forceinline__ float tanh_fast(float x) {
    x = fminf(15.0f, fmaxf(-15.0f, x));
    float e = __expf(2.0f * x);
    return (e - 1.0f) / (e + 1.0f);
}

// ---------------- grid-wide barrier (generation based) ----------------
__device__ __forceinline__ void grid_sync(unsigned* barc, unsigned& gen) {
    __threadfence();            // release this thread's global writes (agent scope)
    __syncthreads();
    if (threadIdx.x == 0) {
        unsigned g   = gen;
        unsigned old = __hip_atomic_fetch_add(&barc[0], 1u, __ATOMIC_ACQ_REL,
                                              __HIP_MEMORY_SCOPE_AGENT);
        if (old == (g + 1u) * NBLK - 1u) {
            __hip_atomic_store(&barc[1], g + 1u, __ATOMIC_RELEASE,
                               __HIP_MEMORY_SCOPE_AGENT);
        } else {
            while (__hip_atomic_load(&barc[1], __ATOMIC_ACQUIRE,
                                     __HIP_MEMORY_SCOPE_AGENT) < g + 1u) {
                __builtin_amdgcn_s_sleep(8);
            }
        }
    }
    __syncthreads();
    gen += 1u;
    __threadfence();            // acquire for all threads
}

// ---------------- fragment loaders ----------------
// A-frag (16x32 f16) ISA layout: lane holds row M = lane%16,
// K(lane,h) = 32*kt + 16*(h/8) + 8*(lane/16) + (h%8)

// from f32 tensor x[B=64][S=1024][F=512], on-the-fly cvt
__device__ __forceinline__ v16h load_x_frag(const float* __restrict__ x,
                                            int t, int mt, int kt, int lane) {
    int r     = mt * 16 + (lane & 15);
    int kbase = kt * 32 + ((lane >> 4) << 3);
    const float* p = x + ((size_t)r * 1024 + t) * 512 + kbase;
    float4 f0 = *(const float4*)(p);
    float4 f1 = *(const float4*)(p + 4);
    float4 f2 = *(const float4*)(p + 16);
    float4 f3 = *(const float4*)(p + 20);
    v16h a;
    a[0]=(half_t)f0.x; a[1]=(half_t)f0.y; a[2]=(half_t)f0.z; a[3]=(half_t)f0.w;
    a[4]=(half_t)f1.x; a[5]=(half_t)f1.y; a[6]=(half_t)f1.z; a[7]=(half_t)f1.w;
    a[8]=(half_t)f2.x; a[9]=(half_t)f2.y; a[10]=(half_t)f2.z; a[11]=(half_t)f2.w;
    a[12]=(half_t)f3.x; a[13]=(half_t)f3.y; a[14]=(half_t)f3.z; a[15]=(half_t)f3.w;
    return a;
}

// from row-major f16 h buffer [64][768]
__device__ __forceinline__ v16h load_h_frag(const half_t* __restrict__ hb,
                                            int mt, int kt, int lane) {
    int r     = mt * 16 + (lane & 15);
    int kbase = kt * 32 + ((lane >> 4) << 3);
    const half_t* p = hb + (size_t)r * 768 + kbase;
    v8h g1 = *(const v8h*)(p);
    v8h g2 = *(const v8h*)(p + 16);
    return __builtin_shufflevector(g1, g2, 0,1,2,3,4,5,6,7,8,9,10,11,12,13,14,15);
}

// one-hot(pred_prev) rows, classes kt2*32 .. kt2*32+31
__device__ __forceinline__ v16h onehot_frag(const int* __restrict__ predprev,
                                            int mt, int kt2, int lane) {
    int r    = mt * 16 + (lane & 15);
    int pv   = predprev[r];
    int base = kt2 * 32 + ((lane >> 4) << 3);
    v16h a;
#pragma unroll
    for (int h = 0; h < 16; ++h) {
        int cc = base + ((h >> 3) << 4) + (h & 7);
        a[h] = (pv == cc) ? (half_t)1.0f : (half_t)0.0f;
    }
    return a;
}

// B-frag (32x16 f16) from pre-packed weights: one contiguous 32B load per lane
__device__ __forceinline__ v16h load_b_frag(const v16h* __restrict__ pw,
                                            int ntile, int kt, int KT, int lane) {
    return pw[((size_t)ntile * KT + kt) * 32 + lane];
}

// accumulate one A-fragment into the 4 gate accumulators
__device__ __forceinline__ void gates_accum(v8f acc[4], v16h a,
                                            const v16h* __restrict__ pw,
                                            int nt, int kt, int KT, int lane) {
#pragma unroll
    for (int g = 0; g < 4; ++g) {
        v16h b = load_b_frag(pw, g * 48 + nt, kt, KT, lane);
        acc[g] = __builtin_amdgcn_wmma_f32_16x16x32_f16(
            false, a, false, b, (short)0, acc[g], false, false);
    }
}

// LSTM elementwise + store h tile (C-frag layout: row = v + 8*(lane/16), col = lane%16)
__device__ __forceinline__ void lstm_ew_store(v8f acc[4], float bi, float bf,
                                              float bg, float bo, float cv[8],
                                              half_t* __restrict__ hout,
                                              int mt, int ncol, int lh) {
#pragma unroll
    for (int v = 0; v < 8; ++v) {
        float iv = sigm(acc[0][v] + bi);
        float fv = sigm(acc[1][v] + bf);
        float gv = tanh_fast(acc[2][v] + bg);
        float ov = sigm(acc[3][v] + bo);
        cv[v] = fv * cv[v] + iv * gv;
        float hv = ov * tanh_fast(cv[v]);
        hout[(size_t)(mt * 16 + v + 8 * lh) * 768 + ncol] = (half_t)hv;
    }
}

// ---------------- weight pack: f32 [N][K] -> f16 WMMA B-fragments ----------------
// layout: dst[((ntile*KT + ktile)*32 + lane)*16 + h],
//         n = ntile*16 + lane%16 ; k = ktile*32 + 8*(lane/16) + 16*(h/8) + h%8
__global__ void pack_w_kernel(const float* __restrict__ src,
                              half_t* __restrict__ dst, int N, int K) {
    long long p     = (long long)blockIdx.x * 256 + threadIdx.x;
    long long total = (long long)N * K;
    if (p >= total) return;
    int h       = (int)(p & 15);
    int lane    = (int)((p >> 4) & 31);
    long long tile = p >> 9;
    int KT      = K >> 5;
    int ktile   = (int)(tile % KT);
    int ntile   = (int)(tile / KT);
    int n = ntile * 16 + (lane & 15);
    int k = ktile * 32 + ((lane >> 4) << 3) + ((h >> 3) << 4) + (h & 7);
    dst[p] = (half_t)src[(size_t)n * K + k];
}

// ---------------- persistent decoder kernel ----------------
__global__ __launch_bounds__(256)
void lstm_decoder_kernel(const float* __restrict__ x, const int* __restrict__ tag,
                         const float* __restrict__ b0, const float* __restrict__ b1,
                         const float* __restrict__ blin,
                         const half_t* __restrict__ PWih0, const half_t* __restrict__ PWhh0,
                         const half_t* __restrict__ PWih1, const half_t* __restrict__ PWhh1,
                         const half_t* __restrict__ PWlin,
                         half_t* __restrict__ h0buf, half_t* __restrict__ h1buf,
                         int* __restrict__ predbuf, float* __restrict__ logits,
                         unsigned* __restrict__ barc, float* __restrict__ out) {
    const int tid  = threadIdx.x;
    const int lane = tid & 31;
    const int wave = tid >> 5;
    const int gw   = blockIdx.x * 8 + wave;  // 0..191
    const int mt   = gw / 48;                // batch tile 0..3
    const int nt   = gw % 48;                // hidden tile 0..47
    const int lr   = lane & 15;
    const int lh   = lane >> 4;
    const int ncol = nt * 16 + lr;

    const v16h* pwih0 = (const v16h*)PWih0;  // KT = 18
    const v16h* pwhh0 = (const v16h*)PWhh0;  // KT = 24
    const v16h* pwih1 = (const v16h*)PWih1;  // KT = 24
    const v16h* pwhh1 = (const v16h*)PWhh1;  // KT = 24
    const v16h* pwlin = (const v16h*)PWlin;  // KT = 24

    // hoist biases (fixed per wave/lane)
    const float bi0 = b0[ncol],            bf0 = b0[768 + ncol];
    const float bg0 = b0[1536 + ncol],     bo0 = b0[2304 + ncol];
    const float bi1 = b1[ncol],            bf1 = b1[768 + ncol];
    const float bg1 = b1[1536 + ncol],     bo1 = b1[2304 + ncol];

    float c0v[8], c1v[8];
#pragma unroll
    for (int v = 0; v < 8; ++v) { c0v[v] = 0.f; c1v[v] = 0.f; }

    float    loss_b = 0.f;
    unsigned gen    = 0;

    for (int t = 0; t < 1024; ++t) {
        const int par  = t & 1;
        const int parp = par ^ 1;

        // ---------------- layer 0 ----------------
        {
            v8f acc[4] = { {0,0,0,0,0,0,0,0}, {0,0,0,0,0,0,0,0},
                           {0,0,0,0,0,0,0,0}, {0,0,0,0,0,0,0,0} };
            // x part: K = 0..511 (k-tiles 0..15 of W_ih0)
            for (int kt = 0; kt < 16; ++kt) {
                v16h a = load_x_frag(x, t, mt, kt, lane);
                gates_accum(acc, a, pwih0, nt, kt, 18, lane);
            }
            // one-hot(pred) part: K = 512..575 (k-tiles 16..17)
            const int* pp = predbuf + parp * 64;
#pragma unroll
            for (int kt2 = 0; kt2 < 2; ++kt2) {
                v16h a = onehot_frag(pp, mt, kt2, lane);
                gates_accum(acc, a, pwih0, nt, 16 + kt2, 18, lane);
            }
            // recurrent part: h0(t-1)
            const half_t* h0p = h0buf + (size_t)parp * BH;
            for (int kt = 0; kt < 24; ++kt) {
                if (kt + 1 < 24)  // warm next B-frags (L2 resident weights -> WGP$)
                    __builtin_prefetch((const void*)(pwhh0 + ((size_t)nt * 24 + kt + 1) * 32 + lane), 0, 1);
                v16h a = load_h_frag(h0p, mt, kt, lane);
                gates_accum(acc, a, pwhh0, nt, kt, 24, lane);
            }
            lstm_ew_store(acc, bi0, bf0, bg0, bo0, c0v,
                          h0buf + (size_t)par * BH, mt, ncol, lh);
        }
        grid_sync(barc, gen);   // h0(t) visible

        // ---------------- layer 1 ----------------
        {
            v8f acc[4] = { {0,0,0,0,0,0,0,0}, {0,0,0,0,0,0,0,0},
                           {0,0,0,0,0,0,0,0}, {0,0,0,0,0,0,0,0} };
            const half_t* h0c = h0buf + (size_t)par * BH;
            for (int kt = 0; kt < 24; ++kt) {
                v16h a = load_h_frag(h0c, mt, kt, lane);
                gates_accum(acc, a, pwih1, nt, kt, 24, lane);
            }
            const half_t* h1p = h1buf + (size_t)parp * BH;
            for (int kt = 0; kt < 24; ++kt) {
                v16h a = load_h_frag(h1p, mt, kt, lane);
                gates_accum(acc, a, pwhh1, nt, kt, 24, lane);
            }
            lstm_ew_store(acc, bi1, bf1, bg1, bo1, c1v,
                          h1buf + (size_t)par * BH, mt, ncol, lh);
        }
        grid_sync(barc, gen);   // h1(t) visible

        // ---------------- logits + softmax/NLL/argmax (block 0 only) ----------------
        if (blockIdx.x == 0) {
            const half_t* h1c = h1buf + (size_t)par * BH;
#pragma unroll
            for (int q = 0; q < 2; ++q) {        // 2 logit tiles per wave (16 total)
                int lt  = wave * 2 + q;
                int lmt = lt >> 2, lnt = lt & 3;
                v8f acc = {0,0,0,0,0,0,0,0};
                for (int kt = 0; kt < 24; ++kt) {
                    v16h a = load_h_frag(h1c, lmt, kt, lane);
                    v16h b = load_b_frag(pwlin, lnt, kt, 24, lane);
                    acc = __builtin_amdgcn_wmma_f32_16x16x32_f16(
                        false, a, false, b, (short)0, acc, false, false);
                }
                float bl = blin[lnt * 16 + lr];
#pragma unroll
                for (int v = 0; v < 8; ++v)
                    logits[(size_t)(lmt * 16 + v + 8 * lh) * 64 + lnt * 16 + lr] = acc[v] + bl;
            }
            __threadfence_block();
            __syncthreads();
            if (tid < 64) {
                int   b  = tid;
                float mx = -3.4e38f;
                int   am = 0;
                for (int c = 0; c < 64; ++c) {
                    float v = logits[(size_t)b * 64 + c];
                    if (v > mx) { mx = v; am = c; }
                }
                float se = 0.f;
                for (int c = 0; c < 64; ++c)
                    se += __expf(logits[(size_t)b * 64 + c] - mx);
                float lse = mx + __logf(se);
                int tt = tag[(size_t)b * 1024 + t];
                if (tt != 0) loss_b += (lse - logits[(size_t)b * 64 + tt]);
                predbuf[par * 64 + b] = am;
            }
        }
        grid_sync(barc, gen);   // pred(t) visible
    }

    if (blockIdx.x == 0) {
        __shared__ float red[64];
        if (tid < 64) red[tid] = loss_b;
        __syncthreads();
        if (tid == 0) {
            float s = 0.f;
            for (int i = 0; i < 64; ++i) s += red[i];
            out[0] = s;
        }
    }
}

// ---------------- host launcher ----------------
extern "C" void kernel_launch(void* const* d_in, const int* in_sizes, int n_in,
                              void* d_out, int out_size, void* d_ws, size_t ws_size,
                              hipStream_t stream) {
    (void)in_sizes; (void)n_in; (void)out_size; (void)ws_size;
    const float* x    = (const float*)d_in[0];
    const int*   tag  = (const int*)d_in[1];
    const float* Wih0 = (const float*)d_in[2];
    const float* Whh0 = (const float*)d_in[3];
    const float* b0   = (const float*)d_in[4];
    const float* Wih1 = (const float*)d_in[5];
    const float* Whh1 = (const float*)d_in[6];
    const float* b1   = (const float*)d_in[7];
    const float* Wlin = (const float*)d_in[8];
    const float* blin = (const float*)d_in[9];

    char*  ws  = (char*)d_ws;
    size_t off = 0;
    auto alloc = [&](size_t bytes) -> void* {
        void* p = ws + off;
        off += (bytes + 255) & ~(size_t)255;
        return p;
    };
    unsigned* barc    = (unsigned*)alloc(256);                // [arrive, gen]
    int*      predbuf = (int*)     alloc(2 * 64 * sizeof(int));
    half_t*   h0buf   = (half_t*)  alloc(2 * BH * sizeof(half_t));
    half_t*   h1buf   = (half_t*)  alloc(2 * BH * sizeof(half_t));
    float*    logits  = (float*)   alloc(64 * 64 * sizeof(float));
    size_t ctrl_bytes = off;                                  // zero each launch
    half_t* PWih0 = (half_t*)alloc((size_t)3072 * 576 * sizeof(half_t));
    half_t* PWhh0 = (half_t*)alloc((size_t)3072 * 768 * sizeof(half_t));
    half_t* PWih1 = (half_t*)alloc((size_t)3072 * 768 * sizeof(half_t));
    half_t* PWhh1 = (half_t*)alloc((size_t)3072 * 768 * sizeof(half_t));
    half_t* PWlin = (half_t*)alloc((size_t)64 * 768 * sizeof(half_t));

    hipMemsetAsync(d_ws, 0, ctrl_bytes, stream);  // zero h(-1), pred(-1), barrier

    auto packs = [&](const float* src, half_t* dst, int N, int K) {
        long long total = (long long)N * K;
        int blocks = (int)((total + 255) / 256);
        pack_w_kernel<<<blocks, 256, 0, stream>>>(src, dst, N, K);
    };
    packs(Wih0, PWih0, 3072, 576);
    packs(Whh0, PWhh0, 3072, 768);
    packs(Wih1, PWih1, 3072, 768);
    packs(Whh1, PWhh1, 3072, 768);
    packs(Wlin, PWlin, 64, 768);

    lstm_decoder_kernel<<<NBLK, 256, 0, stream>>>(
        x, tag, b0, b1, blin, PWih0, PWhh0, PWih1, PWhh1, PWlin,
        h0buf, h1buf, predbuf, logits, barc, (float*)d_out);
}
